// Involution_75943611728393
// MI455X (gfx1250) — matrix operand order repaired and checked by reading, not compiled, not run on hardware
//
#include <hip/hip_runtime.h>
#include <hip/hip_bf16.h>

typedef __attribute__((ext_vector_type(16))) _Float16 v16h;
typedef __attribute__((ext_vector_type(8)))  float    v8f;
typedef unsigned int tdm_v4u __attribute__((ext_vector_type(4)));
typedef int          tdm_v4i __attribute__((ext_vector_type(4)));
typedef int          tdm_v8i __attribute__((ext_vector_type(8)));

#define KS   7
#define KK   49          // KS*KS
#define CIN  256
#define CR   64          // CIN/4
#define GRP  16          // CIN/16
#define HH   56
#define WW   56
#define BB   4
#define OUTC 784         // KK*GRP
#define EPSB 1e-5f

// pixel tile handled by one wave: 2 rows x 8 cols = 16 pixels (WMMA N dim)
#define TH_  2
#define TW_  8
#define TPW  (WW / TW_)  // 7
#define TPH  (HH / TH_)  // 28

// x halo staging: 64-channel chunks, window <= 8 rows x 14 cols
#define NCHK      64
#define CHUNK_FLT (NCHK * 8 * 14)   // 6272 floats = 25088 B per buffer

#define A1_HALVES (4 * 8 * 32 * 16)
#define A2_HALVES (49 * 2 * 32 * 16)

#if defined(__has_builtin)
#  if __has_builtin(__builtin_amdgcn_tensor_load_to_lds)
#    define HAVE_TDM 1
#  endif
#endif
#ifndef HAVE_TDM
#  define HAVE_TDM 0
#endif

// Contiguous A-matrix (16x32 f16) per-lane K index (k = 16*(L/16) + e): element e of lane L
__device__ __forceinline__ int a_kidx(int lane, int e) {
    return e + ((lane >= 16) ? 16 : 0);
}

// ---------------------------------------------------------------------------
// TDM: 3-D tile load (x=cols, y=rows, z=channels) Global -> LDS.
// D# packing per CDNA5 ISA (§8.3-8.6).
// ---------------------------------------------------------------------------
__device__ __forceinline__ void chunk_load(float* dst_lds, const float* src,
                                           int nr, int nc, int lane) {
#if HAVE_TDM
    (void)lane;
    unsigned long long ga = (unsigned long long)(uintptr_t)src;
    unsigned lds_off = (unsigned)(uintptr_t)dst_lds;   // low 32b = LDS byte addr
    tdm_v4u g0 = {
        1u,                                            // count=1, user D#
        lds_off,                                       // lds_addr
        (unsigned)(ga & 0xFFFFFFFFu),                  // global_addr[31:0]
        (unsigned)((ga >> 32) & 0x1FFFFFFu) | (2u << 30) // addr[56:32], type=2
    };
    tdm_v8i g1 = {
        (int)(2u << 16),                               // data_size = 4 bytes
        (int)((unsigned)(nc & 0xFFFF) << 16),          // tensor_dim0[15:0]
        (int)(((unsigned)nr & 0xFFFF) << 16),          // dim0 hi=0 | tensor_dim1 lo
        (int)(((unsigned)nc & 0xFFFF) << 16),          // dim1 hi=0 | tile_dim0 = nc
        (int)(((unsigned)nr & 0xFFFF) | ((unsigned)NCHK << 16)), // tile_dim1, tile_dim2
        (int)(WW),                                     // tensor_dim0_stride = 56
        (int)(((unsigned)(HH * WW) & 0xFFFF) << 16),   // stride0 hi=0 | stride1[15:0]
        (int)((unsigned)(HH * WW) >> 16)               // tensor_dim1_stride[47:16]
    };
    tdm_v4i g2 = { NCHK, 0, 0, 0 };                    // tensor_dim2 = 64
    tdm_v4i g3 = { 0, 0, 0, 0 };
#if defined(__clang_major__) && (__clang_major__ >= 23)
    tdm_v8i gz = { 0, 0, 0, 0, 0, 0, 0, 0 };
    __builtin_amdgcn_tensor_load_to_lds(g0, g1, g2, g3, gz, 0);
#else
    __builtin_amdgcn_tensor_load_to_lds(g0, g1, g2, g3, 0);
#endif
#else
    int tot = NCHK * nr * nc, rs = nr * nc;
    for (int i = lane; i < tot; i += 32) {
        int z = i / rs, rem = i % rs;
        dst_lds[i] = src[(size_t)z * (HH * WW) + (rem / nc) * WW + (rem % nc)];
    }
#endif
}

__device__ __forceinline__ void chunk_wait_one(void) {
#if HAVE_TDM
    __builtin_amdgcn_s_wait_tensorcnt(1);
#endif
}
__device__ __forceinline__ void chunk_wait_zero(void) {
#if HAVE_TDM
    __builtin_amdgcn_s_wait_tensorcnt(0);
#endif
}

// ---------------------------------------------------------------------------
// Prep: fold BN into conv1, pre-pack conv1/conv2 into WMMA A-operand layout.
// ---------------------------------------------------------------------------
__global__ void invol_prep(const float* __restrict__ conv1_w,
                           const float* __restrict__ bn_g,
                           const float* __restrict__ bn_b,
                           const float* __restrict__ bn_m,
                           const float* __restrict__ bn_v,
                           const float* __restrict__ conv2_w,
                           _Float16* __restrict__ A1,
                           _Float16* __restrict__ A2,
                           float* __restrict__ bias1) {
    int idx = blockIdx.x * blockDim.x + threadIdx.x;

    if (idx < CR) {
        float s = bn_g[idx] * rsqrtf(bn_v[idx] + EPSB);
        bias1[idx] = bn_b[idx] - bn_m[idx] * s;
    }
    if (idx < A1_HALVES) {
        int e = idx & 15, lane = (idx >> 4) & 31, kc = (idx >> 9) & 7, mt = idx >> 12;
        int M = mt * 16 + (lane & 15);
        int Kc = kc * 32 + a_kidx(lane, e);
        float s = bn_g[M] * rsqrtf(bn_v[M] + EPSB);
        A1[idx] = (_Float16)(conv1_w[M * CIN + Kc] * s);
    }
    if (idx < A2_HALVES) {
        int e = idx & 15, lane = (idx >> 4) & 31, kc = (idx >> 9) & 1, mt = idx >> 10;
        int M = mt * 16 + (lane & 15);
        int Kc = kc * 32 + a_kidx(lane, e);
        A2[idx] = (_Float16)(conv2_w[M * CR + Kc]);
    }
}

// ---------------------------------------------------------------------------
// Fused involution: TDM prefetch of x halo || GEMM1 (WMMA) -> BN/ReLU ->
// GEMM2 (WMMA, kernels in LDS) -> 7x7 weighted reduce from LDS.
// One wave (32 threads) per 16-pixel tile.
// ---------------------------------------------------------------------------
__global__ __launch_bounds__(32) void invol_main(
        const float* __restrict__ x,
        const float* __restrict__ bias2,       // conv2_b (784)
        const _Float16* __restrict__ A1p,
        const _Float16* __restrict__ A2p,
        const float* __restrict__ bias1,
        float* __restrict__ out) {
    __shared__ _Float16 sW1[CR * 16];          //  2 KB
    __shared__ float    sW2[OUTC * 16];        // 50 KB
    __shared__ float    sX[2 * CHUNK_FLT];     // 50 KB (double-buffered halo)

    const int lane = threadIdx.x;              // wave32
    const int n    = lane & 15;                // column / pixel index
    const int hi   = lane >> 4;

    int blk = blockIdx.x;
    int b   = blk / (TPW * TPH);
    int rem = blk % (TPW * TPH);
    int h0  = (rem / TPW) * TH_;
    int w0  = (rem % TPW) * TW_;
    int ph  = h0 + (n >> 3);
    int pw  = w0 + (n & 7);
    int pixOff = ph * WW + pw;
    const float* xb = x + (size_t)b * CIN * HH * WW;

    // clamped halo window for this tile (rows [r0,r1], cols [c0,c1])
    int r0 = h0 - 3;  if (r0 < 0) r0 = 0;
    int r1 = h0 + 4;  if (r1 > HH - 1) r1 = HH - 1;
    int c0 = w0 - 3;  if (c0 < 0) c0 = 0;
    int c1 = w0 + 10; if (c1 > WW - 1) c1 = WW - 1;
    int nr = r1 - r0 + 1, nc = c1 - c0 + 1;
    int tileN = nr * nc;
    const float* xwin = xb + r0 * WW + c0;

    // kick off async halo DMA for channel chunks 0,1 (overlaps the GEMMs)
    chunk_load(&sX[0],         xwin,                      nr, nc, lane);
    chunk_load(&sX[CHUNK_FLT], xwin + (size_t)NCHK * HH * WW, nr, nc, lane);

    // ---- GEMM1: w1(64x16) = A1fold(64x256) x xtile(256x16) ----
    const v16h* a1 = (const v16h*)A1p;
    v8f acc[4] = {v8f{}, v8f{}, v8f{}, v8f{}};
    #pragma unroll
    for (int kc = 0; kc < 8; ++kc) {
        v16h bv;
        int cbase = kc * 32 + hi * 16;
        #pragma unroll
        for (int e = 0; e < 16; ++e)
            bv[e] = (_Float16)xb[(cbase + e) * (HH * WW) + pixOff];
        #pragma unroll
        for (int mt = 0; mt < 4; ++mt) {
            v16h av = a1[(mt * 8 + kc) * 32 + lane];
            acc[mt] = __builtin_amdgcn_wmma_f32_16x16x32_f16(
                false, av, false, bv, (short)0, acc[mt], false, false);
        }
    }
    #pragma unroll
    for (int mt = 0; mt < 4; ++mt)
        #pragma unroll
        for (int r = 0; r < 8; ++r) {
            int M = mt * 16 + r + hi * 8;
            float v = acc[mt][r] + bias1[M];
            sW1[M * 16 + n] = (_Float16)fmaxf(v, 0.0f);
        }
    __syncthreads();

    v16h b20, b21;
    #pragma unroll
    for (int e = 0; e < 16; ++e) {
        b20[e] = sW1[(hi * 16 + e) * 16 + n];
        b21[e] = sW1[(32 + hi * 16 + e) * 16 + n];
    }

    // ---- GEMM2: w2(784x16) = conv2(784x64) x w1(64x16) + b -> LDS ----
    const v16h* a2 = (const v16h*)A2p;
    for (int mt = 0; mt < KK; ++mt) {
        v16h aA = a2[(mt * 2 + 0) * 32 + lane];
        v16h aB = a2[(mt * 2 + 1) * 32 + lane];
        v8f c = {};
        c = __builtin_amdgcn_wmma_f32_16x16x32_f16(false, aA, false, b20, (short)0, c, false, false);
        c = __builtin_amdgcn_wmma_f32_16x16x32_f16(false, aB, false, b21, (short)0, c, false, false);
        #pragma unroll
        for (int r = 0; r < 8; ++r) {
            int M2 = mt * 16 + r + hi * 8;
            sW2[M2 * 16 + n] = c[r] + bias2[M2];
        }
    }
    __syncthreads();

    // ---- Stage 3: out[c,p] = sum_k w2[g*49+k, p] * x[c, p + delta_k] ----
    // channel chunks of 64, double-buffered TDM pipeline
    #pragma unroll
    for (int cc = 0; cc < 4; ++cc) {
        if (cc < 3) chunk_wait_one();
        else        chunk_wait_zero();
        __syncthreads();
        const float* xs = &sX[(cc & 1) * CHUNK_FLT];
        for (int it = 0; it < 32; ++it) {
            int zc = it * 2 + hi;                     // channel within chunk
            int c  = cc * NCHK + zc;
            int g  = c >> 4;
            const float* wrow = &sW2[g * KK * 16 + n];
            const float* xz = xs + zc * tileN;
            float s = 0.0f;
            #pragma unroll
            for (int kh = 0; kh < KS; ++kh) {
                int y = ph + kh - 3 - r0;
                bool oky = (unsigned)y < (unsigned)nr;
                #pragma unroll
                for (int kw = 0; kw < KS; ++kw) {
                    int xx = pw + kw - 3 - c0;
                    float xv = (oky && (unsigned)xx < (unsigned)nc)
                                   ? xz[y * nc + xx] : 0.0f;
                    s = fmaf(wrow[(kh * KS + kw) * 16], xv, s);
                }
            }
            out[((size_t)(b * CIN + c) * HH + ph) * WW + pw] = s;
        }
        __syncthreads();
        if (cc + 2 < 4)
            chunk_load(&sX[(cc & 1) * CHUNK_FLT],
                       xwin + (size_t)(cc + 2) * NCHK * HH * WW, nr, nc, lane);
    }
}

extern "C" void kernel_launch(void* const* d_in, const int* in_sizes, int n_in,
                              void* d_out, int out_size, void* d_ws, size_t ws_size,
                              hipStream_t stream) {
    (void)in_sizes; (void)n_in; (void)out_size; (void)ws_size;
    const float* x       = (const float*)d_in[0];
    const float* conv1_w = (const float*)d_in[1];
    const float* bn_g    = (const float*)d_in[2];
    const float* bn_b    = (const float*)d_in[3];
    const float* bn_m    = (const float*)d_in[4];
    const float* bn_v    = (const float*)d_in[5];
    const float* conv2_w = (const float*)d_in[6];
    const float* conv2_b = (const float*)d_in[7];
    float* out = (float*)d_out;

    _Float16* A1 = (_Float16*)d_ws;
    _Float16* A2 = A1 + A1_HALVES;
    float* bias1 = (float*)(A2 + A2_HALVES);

    invol_prep<<<(A2_HALVES + 255) / 256, 256, 0, stream>>>(
        conv1_w, bn_g, bn_b, bn_m, bn_v, conv2_w, A1, A2, bias1);

    invol_main<<<BB * TPH * TPW, 32, 0, stream>>>(
        x, conv2_b, A1, A2, bias1, out);
}
